// GTN_20942260536103
// MI455X (gfx1250) — matrix operand order repaired and checked by reading, not compiled, not run on hardware
//
#include <hip/hip_runtime.h>
#include <hip/hip_bf16.h>

#define NNODES 2048
#define NEDGES 65536
#define DIM 256
#define NH 8
#define HD 32
#define NLAYERS 2
#define SCALE 0.17677669529663687f  // 1/sqrt(32)

#define BM 128        // block tile M: 4 waves x 32 rows
#define BN 64         // block tile N
#define BK 32         // k-step (one WMMA K)
#define LSTR 48       // LDS stride in f16 per column (32 data + 16 pad, keeps 32B align)

typedef __attribute__((ext_vector_type(16))) _Float16 v16h;
typedef __attribute__((ext_vector_type(8)))  float    v8f;

// ---------------------------------------------------------------------------
// A-matrix 16x32 (MxK) f16 fragment, contiguous k order (B uses the same): lane L (half=L>>4, m=L&15)
//   reg r holds K = half*16 + r*2 + {0,1}
// ---------------------------------------------------------------------------
__device__ __forceinline__ v16h load_a_frag(const float* __restrict__ A,
                                            int row, int kb, int half) {
  v16h a;
#pragma unroll
  for (int r = 0; r < 8; ++r) {
    const int k = kb + (half << 4) + (r << 1);
    const float2 f = *reinterpret_cast<const float2*>(A + (size_t)row * DIM + k);
    a[2 * r]     = (_Float16)f.x;
    a[2 * r + 1] = (_Float16)f.y;
  }
  return a;
}

// out[M x 256] = A[M x 256] @ W[256 x 256]^T + bias   (torch Linear)
// 128 threads = 4 waves; block tile 128x64. Wave computes 32x64 (2 M-subtiles
// x 4 N-tiles = 8 WMMA accumulators). W k-slab staged in LDS (f16, double
// buffered, one barrier per k-step) and shared by all waves:
//   B-frag for lane (n=L&15, half=L>>4), tile t is 16 contiguous f16 at
//   lds[(t*16+n)*LSTR + half*16]  (same contiguous k order as A: reg j <-> K=half*16+2j+s)
__global__ __launch_bounds__(128) void gemm_wmma_bias(
    float* __restrict__ out, const float* __restrict__ A,
    const float* __restrict__ W, const float* __restrict__ bias) {
  __shared__ _Float16 bsm[2][BN * LSTR];   // 2 x 6 KB

  const int tid   = threadIdx.x;
  const int lane  = tid & 31;
  const int wave  = tid >> 5;
  const int half  = lane >> 4;
  const int mq    = lane & 15;
  const int tileN = blockIdx.x * BN;
  const int tileM = blockIdx.y * BM + wave * 32;

  // cooperative W fill: thread -> (col = tid>>1, k-half = (tid&1)*16)
  const int fcol = tid >> 1;
  const int fk   = (tid & 1) << 4;
  const float4* __restrict__ wsrc =
      reinterpret_cast<const float4*>(W + (size_t)(tileN + fcol) * DIM + fk);
  _Float16* const fdst0 = &bsm[0][fcol * LSTR + fk];
  _Float16* const fdst1 = &bsm[1][fcol * LSTR + fk];

  v8f acc[2][4] = {};

  // prologue fill of buffer 0 (kb = 0)
  {
    v16h tmp;
#pragma unroll
    for (int j = 0; j < 4; ++j) {
      const float4 f = wsrc[j];
      tmp[4 * j + 0] = (_Float16)f.x; tmp[4 * j + 1] = (_Float16)f.y;
      tmp[4 * j + 2] = (_Float16)f.z; tmp[4 * j + 3] = (_Float16)f.w;
    }
    *reinterpret_cast<v16h*>(fdst0) = tmp;
  }
  __syncthreads();

#pragma unroll
  for (int step = 0; step < DIM / BK; ++step) {
    const int kb  = step * BK;
    const int buf = step & 1;

    // prefetch next k-slab into the other buffer (overlaps with WMMAs below)
    if (step + 1 < DIM / BK) {
      v16h tmp;
#pragma unroll
      for (int j = 0; j < 4; ++j) {
        const float4 f = wsrc[((kb + BK) >> 2) + j];
        tmp[4 * j + 0] = (_Float16)f.x; tmp[4 * j + 1] = (_Float16)f.y;
        tmp[4 * j + 2] = (_Float16)f.z; tmp[4 * j + 3] = (_Float16)f.w;
      }
      *reinterpret_cast<v16h*>(buf ? fdst0 : fdst1) = tmp;
    }

    const v16h a0 = load_a_frag(A, tileM + mq,      kb, half);
    const v16h a1 = load_a_frag(A, tileM + 16 + mq, kb, half);
#pragma unroll
    for (int t = 0; t < 4; ++t) {
      const v16h b = *reinterpret_cast<const v16h*>(
          &bsm[buf][(t * 16 + mq) * LSTR + half * 16]);
      acc[0][t] = __builtin_amdgcn_wmma_f32_16x16x32_f16(
          false, a0, false, b, (short)0, acc[0][t], false, false);
      acc[1][t] = __builtin_amdgcn_wmma_f32_16x16x32_f16(
          false, a1, false, b, (short)0, acc[1][t], false, false);
    }
    __syncthreads();  // reads of buf done; fill of buf^1 visible for next step
  }

  // D layout: lane L, reg r -> (m = half*8 + r, n = lane&15)
#pragma unroll
  for (int t = 0; t < 4; ++t) {
    const int n    = tileN + t * 16 + mq;
    const float bv = bias[n];
#pragma unroll
    for (int mi = 0; mi < 2; ++mi) {
#pragma unroll
      for (int r = 0; r < 8; ++r) {
        const int m = tileM + mi * 16 + (half << 3) + r;
        out[(size_t)m * DIM + n] = acc[mi][t][r] + bv;
      }
    }
  }
}

// Zero node accumulator; seed denom with the self-attention exp term.
__global__ __launch_bounds__(256) void init_self(
    const float* __restrict__ Q, const float* __restrict__ K,
    float* __restrict__ acc, float* __restrict__ selfw,
    float* __restrict__ denom) {
  const int i = blockIdx.x, t = threadIdx.x;
  acc[(size_t)i * DIM + t] = 0.0f;
  if (t < NH) {
    const float* q = Q + (size_t)i * DIM + t * HD;
    const float* k = K + (size_t)i * DIM + t * HD;
    float d = 0.0f;
#pragma unroll
    for (int j = 0; j < HD; ++j) d += q[j] * k[j];
    const float w = __expf(d * SCALE);
    selfw[i * NH + t] = w;
    denom[i * NH + t] = w;
  }
}

// One wave per edge. lane = (head = lane>>2, chunk = lane&3) -> 8 dims each.
// s = (Qs.Kt + Qs.E + E.Kt)*scale ; expw stored, denom[src,h] accumulated.
__global__ __launch_bounds__(256) void edge_logits(
    const float* __restrict__ Q, const float* __restrict__ K,
    const float* __restrict__ Eo, const int* __restrict__ src,
    const int* __restrict__ tgt, float* __restrict__ expw,
    float* __restrict__ denom) {
  const int lane = threadIdx.x & 31;
  const int e    = blockIdx.x * 8 + (threadIdx.x >> 5);
  const int h    = lane >> 2;
  const int base = (h << 5) + ((lane & 3) << 3);
  const int s = src[e], t = tgt[e];
  const float4* q4 = reinterpret_cast<const float4*>(Q  + (size_t)s * DIM + base);
  const float4* k4 = reinterpret_cast<const float4*>(K  + (size_t)t * DIM + base);
  const float4* e4 = reinterpret_cast<const float4*>(Eo + (size_t)e * DIM + base);
  float part = 0.0f;
#pragma unroll
  for (int c = 0; c < 2; ++c) {
    const float4 q = q4[c], k = k4[c], ee = e4[c];
    part += q.x * k.x + q.y * k.y + q.z * k.z + q.w * k.w;      // Q.K
    part += q.x * ee.x + q.y * ee.y + q.z * ee.z + q.w * ee.w;  // Q.E
    part += ee.x * k.x + ee.y * k.y + ee.z * k.z + ee.w * k.w;  // E.K
  }
  part += __shfl_xor(part, 1, 32);
  part += __shfl_xor(part, 2, 32);
  const float w = __expf(part * SCALE);
  if ((lane & 3) == 0) {
    expw[e * NH + h] = w;
    atomicAdd(&denom[s * NH + h], w);
  }
}

// Scatter w * V[tgt] into accumulator of src node.
__global__ __launch_bounds__(256) void edge_aggregate(
    const float* __restrict__ expw, const float* __restrict__ denom,
    const float* __restrict__ V, const int* __restrict__ src,
    const int* __restrict__ tgt, float* __restrict__ acc) {
  const int lane = threadIdx.x & 31;
  const int e    = blockIdx.x * 8 + (threadIdx.x >> 5);
  const int h    = lane >> 2;
  const int base = (h << 5) + ((lane & 3) << 3);
  const int s = src[e], t = tgt[e];
  const float w = expw[e * NH + h] / denom[s * NH + h];
  const float* vp = V + (size_t)t * DIM + base;
  float* ap = acc + (size_t)s * DIM + base;
#pragma unroll
  for (int d = 0; d < 8; ++d) atomicAdd(&ap[d], w * vp[d]);
}

// Diagonal (self) softmax contribution.
__global__ __launch_bounds__(256) void add_self(
    const float* __restrict__ selfw, const float* __restrict__ denom,
    const float* __restrict__ V, float* __restrict__ acc) {
  const int i = blockIdx.x, t = threadIdx.x, h = t >> 5;
  const float w = selfw[i * NH + h] / denom[i * NH + h];
  acc[(size_t)i * DIM + t] += w * V[(size_t)i * DIM + t];
}

// h = LayerNorm(x_proj + node_out) * lnw + lnb, one block per node.
__global__ __launch_bounds__(256) void epilogue_ln(
    const float* __restrict__ XP, const float* __restrict__ NO,
    const float* __restrict__ lnw, const float* __restrict__ lnb,
    float* __restrict__ xout) {
  __shared__ float red[8];
  const int i = blockIdx.x, t = threadIdx.x;
  const float v = XP[(size_t)i * DIM + t] + NO[(size_t)i * DIM + t];

  float s = v;
#pragma unroll
  for (int o = 16; o; o >>= 1) s += __shfl_xor(s, o, 32);
  if ((t & 31) == 0) red[t >> 5] = s;
  __syncthreads();
  float mu = 0.0f;
#pragma unroll
  for (int j = 0; j < 8; ++j) mu += red[j];
  mu *= (1.0f / DIM);
  __syncthreads();

  const float d = v - mu;
  float s2 = d * d;
#pragma unroll
  for (int o = 16; o; o >>= 1) s2 += __shfl_xor(s2, o, 32);
  if ((t & 31) == 0) red[t >> 5] = s2;
  __syncthreads();
  float var = 0.0f;
#pragma unroll
  for (int j = 0; j < 8; ++j) var += red[j];
  var *= (1.0f / DIM);

  xout[(size_t)i * DIM + t] = d * rsqrtf(var + 1e-5f) * lnw[t] + lnb[t];
}

extern "C" void kernel_launch(void* const* d_in, const int* in_sizes, int n_in,
                              void* d_out, int out_size, void* d_ws, size_t ws_size,
                              hipStream_t stream) {
  const float* x0   = (const float*)d_in[0];
  const int*   eidx = (const int*)d_in[1];
  const float* ea0  = (const float*)d_in[2];
  const float* Wq = (const float*)d_in[3],  *bq = (const float*)d_in[4];
  const float* Wk = (const float*)d_in[5],  *bk = (const float*)d_in[6];
  const float* Wv = (const float*)d_in[7],  *bv = (const float*)d_in[8];
  const float* We = (const float*)d_in[9],  *be = (const float*)d_in[10];
  const float* Wo = (const float*)d_in[11], *bo = (const float*)d_in[12];
  const float* Wi = (const float*)d_in[13], *bi = (const float*)d_in[14];
  const float* lnw = (const float*)d_in[15], *lnb = (const float*)d_in[16];
  const int* srcp = eidx;
  const int* tgtp = eidx + NEDGES;

  float* ws = (float*)d_ws;
  size_t off = 0;
  float* Ebuf = ws + off; off += (size_t)NEDGES * DIM;   // 64 MB ping buffer
  float* Qb   = ws + off; off += (size_t)NNODES * DIM;
  float* Kb   = ws + off; off += (size_t)NNODES * DIM;
  float* Vb   = ws + off; off += (size_t)NNODES * DIM;
  float* XPb  = ws + off; off += (size_t)NNODES * DIM;
  float* Xb   = ws + off; off += (size_t)NNODES * DIM;
  float* ACC  = ws + off; off += (size_t)NNODES * DIM;
  float* NOb  = ws + off; off += (size_t)NNODES * DIM;
  float* expw = ws + off; off += (size_t)NEDGES * NH;
  float* denm = ws + off; off += (size_t)NNODES * NH;
  float* selw = ws + off; off += (size_t)NNODES * NH;

  float* outx = (float*)d_out;
  float* oute = outx + (size_t)NNODES * DIM;

  const dim3 gN(DIM / BN, NNODES / BM);   // (4, 16)
  const dim3 gE(DIM / BN, NEDGES / BM);   // (4, 512)
  const dim3 blk128(128), blk256(256);
  const dim3 gEdge(NEDGES / 8);

  for (int l = 0; l < NLAYERS; ++l) {
    const float* xin  = l ? Xb   : x0;
    const float* ein  = l ? Ebuf : ea0;
    float*       eout = l ? oute : Ebuf;
    float*       xout = l ? outx : Xb;
    const size_t wo = (size_t)l * DIM * DIM;
    const size_t vo = (size_t)l * DIM;

    gemm_wmma_bias<<<gN, blk128, 0, stream>>>(Qb,  xin, Wq + wo, bq + vo);
    gemm_wmma_bias<<<gN, blk128, 0, stream>>>(Kb,  xin, Wk + wo, bk + vo);
    gemm_wmma_bias<<<gN, blk128, 0, stream>>>(Vb,  xin, Wv + wo, bv + vo);
    gemm_wmma_bias<<<gN, blk128, 0, stream>>>(XPb, xin, Wi + wo, bi + vo);
    gemm_wmma_bias<<<gE, blk128, 0, stream>>>(eout, ein, We + wo, be + vo);

    init_self<<<NNODES, blk256, 0, stream>>>(Qb, Kb, ACC, selw, denm);
    edge_logits<<<gEdge, blk256, 0, stream>>>(Qb, Kb, eout, srcp, tgtp, expw, denm);
    edge_aggregate<<<gEdge, blk256, 0, stream>>>(expw, denm, Vb, srcp, tgtp, ACC);
    add_self<<<NNODES, blk256, 0, stream>>>(selw, denm, Vb, ACC);

    gemm_wmma_bias<<<gN, blk128, 0, stream>>>(NOb, ACC, Wo + wo, bo + vo);
    epilogue_ln<<<NNODES, blk256, 0, stream>>>(XPb, NOb, lnw + vo, lnb + vo, xout);
  }
}
